// AttentionBlock_57612691309421
// MI455X (gfx1250) — matrix operand order repaired and checked by reading, not compiled, not run on hardware
//
#include <hip/hip_runtime.h>

typedef __attribute__((ext_vector_type(16))) _Float16 v16h;
typedef __attribute__((ext_vector_type(8)))  _Float16 v8h;
typedef __attribute__((ext_vector_type(8)))  float    v8f;
typedef __attribute__((ext_vector_type(8)))  unsigned v8u;

#define HW 4096
#define CH 128
#define SK_LD 136   // 32 rows x 136 halves (272B stride: 16B aligned, 4-bank rotate)
#define SV_LD 40    // 128 rows x 40 halves (80B stride: 16B aligned, 20-bank rotate)

static __device__ __forceinline__ v8h ld8(const _Float16* p){ return *(const v8h*)p; }
static __device__ __forceinline__ v16h cat16(v8h a, v8h b){
  return __builtin_shufflevector(a,b,0,1,2,3,4,5,6,7,8,9,10,11,12,13,14,15);
}
// A operand: 16(M) x 32(K) f16 from row-major [rows][ld]; lane<16 holds K k0..k0+15
static __device__ __forceinline__ v16h load_a(const _Float16* base, int row0, int ld, int k0){
  int lane = threadIdx.x & 31; bool low = lane < 16; int r = lane & 15;
  const _Float16* p = base + (size_t)(row0 + r)*ld + k0 + (low ? 0 : 16);
  return cat16(ld8(p), ld8(p+8));
}
// B operand: 32(K) x 16(N) f16; source row-major [N][ld]; lane<16 holds K k0..k0+15 of column n
static __device__ __forceinline__ v16h load_b(const _Float16* base, int row0, int ld, int k0){
  int lane = threadIdx.x & 31; bool low = lane < 16; int r = lane & 15;
  const _Float16* p = base + (size_t)(row0 + r)*ld + k0 + (low ? 0 : 16);
  return cat16(ld8(p), ld8(p+8));
}
static __device__ __forceinline__ v8f wmma_f16(v16h a, v16h b, v8f c){
  return __builtin_amdgcn_wmma_f32_16x16x32_f16(false, a, false, b, (short)0, c, false, false);
}
static __device__ __forceinline__ unsigned pack2(float a, float b){
  unsigned short x = __builtin_bit_cast(unsigned short, (_Float16)a);
  unsigned short y = __builtin_bit_cast(unsigned short, (_Float16)b);
  return (unsigned)x | ((unsigned)y << 16);
}
static __device__ __forceinline__ v8f zero8(){
  v8f z = {0.f,0.f,0.f,0.f,0.f,0.f,0.f,0.f}; return z;
}
// gfx1250 async DMA: copy 16B global -> LDS, tracked by ASYNCcnt
static __device__ __forceinline__ void async_cp16(unsigned lds_off, const _Float16* g){
  asm volatile("global_load_async_to_lds_b128 %0, %1, off"
               :: "v"(lds_off), "v"((unsigned long long)(uintptr_t)g)
               : "memory");
}
static __device__ __forceinline__ void wait_async0(){
  asm volatile("s_wait_asynccnt 0x0" ::: "memory");
}

// ---------------- GroupNorm pass 1: per (b,g) mean / rstd -------------------
__global__ void gn_stats_kernel(const float* __restrict__ x, float* __restrict__ stats){
  __shared__ float s1[256], s2[256];
  int bg = blockIdx.x;                       // b*8+g, group = 16ch*4096 contiguous
  const float* p = x + (size_t)bg * 65536;
  float s = 0.f, ss = 0.f;
  for (int i = threadIdx.x; i < 65536; i += 256){ float v = p[i]; s += v; ss += v*v; }
  s1[threadIdx.x] = s; s2[threadIdx.x] = ss; __syncthreads();
  for (int k = 128; k > 0; k >>= 1){
    if (threadIdx.x < k){ s1[threadIdx.x] += s1[threadIdx.x+k]; s2[threadIdx.x] += s2[threadIdx.x+k]; }
    __syncthreads();
  }
  if (threadIdx.x == 0){
    float mean = s1[0] * (1.0f/65536.0f);
    float var  = s2[0] * (1.0f/65536.0f) - mean*mean;
    stats[bg*2] = mean; stats[bg*2+1] = rsqrtf(var + 1e-5f);
  }
}

// ---------------- GroupNorm pass 2: normalize + transpose to token-major f16 -
__global__ void gn_norm_kernel(const float* __restrict__ x, const float* __restrict__ gamma,
                               const float* __restrict__ beta, const float* __restrict__ stats,
                               _Float16* __restrict__ xnt){
  int bc = blockIdx.x;                       // b*128 + c
  int b = bc >> 7, c = bc & 127, g = c >> 4;
  float mean = stats[(b*8+g)*2], rstd = stats[(b*8+g)*2+1];
  float ga = gamma[c], be = beta[c];
  const float* p = x + (size_t)bc * HW;
  _Float16* q = xnt + (size_t)b*HW*CH + c;
  for (int i = threadIdx.x; i < HW; i += 256){
    float v = (p[i] - mean)*rstd*ga + be;
    q[(size_t)i*CH] = (_Float16)v;
  }
}

// ---------------- weight f32 -> f16 -----------------------------------------
__global__ void cvt_w_kernel(const float* __restrict__ qkvw, const float* __restrict__ projw,
                             _Float16* __restrict__ qw16, _Float16* __restrict__ pw16){
  int i = blockIdx.x*256 + threadIdx.x;      // 65536 = 49152 + 16384
  if (i < 49152) qw16[i] = (_Float16)qkvw[i];
  else           pw16[i-49152] = (_Float16)projw[i-49152];
}

// ---------------- qkv GEMM: [384x128] x [128 x HW] per batch -----------------
__global__ void qkv_gemm_kernel(const _Float16* __restrict__ xnt, const _Float16* __restrict__ qw16,
                                const float* __restrict__ qkvb,
                                _Float16* __restrict__ Q, _Float16* __restrict__ K,
                                _Float16* __restrict__ Vt){
  int w = blockIdx.x*8 + (threadIdx.x >> 5); // 1024 waves: b*256 + tokenTile
  int b = w >> 8; int t0 = (w & 255) * 16;
  int lane = threadIdx.x & 31; bool low = lane < 16; int r16 = lane & 15;
  const _Float16* Bsrc = xnt + (size_t)b*HW*CH;
  v16h bq[4];
  #pragma unroll
  for (int cc = 0; cc < 4; ++cc) bq[cc] = load_b(Bsrc, t0, CH, cc*32);
  const float scale = 0.08838834764831845f;  // 1/sqrt(128), folded into Q
  int token = t0 + r16;
  for (int ot = 0; ot < 24; ++ot){
    v16h a[4];                               // hoist: one clause of 8 loads, one wait
    #pragma unroll
    for (int cc = 0; cc < 4; ++cc) a[cc] = load_a(qw16, ot*16, CH, cc*32);
    v8f acc = zero8();
    #pragma unroll
    for (int cc = 0; cc < 4; ++cc) acc = wmma_f16(a[cc], bq[cc], acc);
    #pragma unroll
    for (int r = 0; r < 8; ++r){
      int o = ot*16 + r + (low ? 0 : 8);
      float v = acc[r] + qkvb[o];
      if (ot < 8)       Q[((size_t)b*HW + token)*CH + o]          = (_Float16)(v*scale);
      else if (ot < 16) K[((size_t)b*HW + token)*CH + (o-128)]    = (_Float16)v;
      else              Vt[(size_t)b*CH*HW + (size_t)(o-256)*HW + token] = (_Float16)v;
    }
  }
}

// ---------------- flash attention -------------------------------------------
// Workgroup = 4 waves x 32 queries (register-blocked: each LDS A-tile feeds
// two WMMAs). K/V key-blocks staged into LDS via gfx1250 async DMA, double-
// buffered against the WMMA pipeline.
__global__ __launch_bounds__(128) void attn_kernel(const _Float16* __restrict__ Q,
                                                   const _Float16* __restrict__ K,
                                                   const _Float16* __restrict__ Vt,
                                                   _Float16* __restrict__ O){
  __shared__ _Float16 sK[2][32*SK_LD];   // token-major K slice
  __shared__ _Float16 sV[2][128*SV_LD];  // channel-major V slice

  int w = blockIdx.x*4 + (threadIdx.x >> 5); // 512 waves: b*128 + queryTile(32q)
  int b = w >> 7; int q0 = (w & 127) * 32;
  int lane = threadIdx.x & 31; bool low = lane < 16; int r16 = lane & 15;
  const _Float16* Qb = Q  + (size_t)b*HW*CH;
  const _Float16* Kb = K  + (size_t)b*HW*CH;
  const _Float16* Vb = Vt + (size_t)b*HW*CH;

  unsigned sK0 = (unsigned)(uintptr_t)&sK[0][0];
  unsigned sK1 = (unsigned)(uintptr_t)&sK[1][0];
  unsigned sV0 = (unsigned)(uintptr_t)&sV[0][0];
  unsigned sV1 = (unsigned)(uintptr_t)&sV[1][0];
  int tid = threadIdx.x;

  // issue async copies of key-block j0 into buffer bi (whole workgroup: 16KB)
  auto issue = [&](int bi, int j0){
    unsigned kbase = bi ? sK1 : sK0;
    unsigned vbase = bi ? sV1 : sV0;
    #pragma unroll
    for (int s = 0; s < 4; ++s){
      int h = (tid + s*128) * 8;               // K: 32x128 halves
      int row = h >> 7, col = h & 127;
      async_cp16(kbase + (unsigned)(row*SK_LD + col)*2, Kb + (size_t)(j0+row)*CH + col);
    }
    #pragma unroll
    for (int s = 0; s < 4; ++s){
      int h = (tid + s*128) * 8;               // V: 128x32 halves
      int c = h >> 5, jj = h & 31;
      async_cp16(vbase + (unsigned)(c*SV_LD + jj)*2, Vb + (size_t)c*HW + j0 + jj);
    }
  };

  // streaming softmax for one 16-query tile: consumes S^T halves dA (keys 0..15)
  // and dB (keys 16..31), updates (m,l), rescales 8 accumulators, returns P^T.
  auto softmax_pack = [&](v8f& dA, v8f& dB, float& m, float& l, v8f* acc) -> v16h {
    float bm = -1e30f;
    #pragma unroll
    for (int r = 0; r < 8; ++r){ bm = fmaxf(bm, dA[r]); bm = fmaxf(bm, dB[r]); }
    bm = fmaxf(bm, __shfl_xor(bm, 16, 32));
    float mn = fmaxf(m, bm);
    float alpha = __expf(m - mn);
    float eA[8], eB[8], ps = 0.f;
    #pragma unroll
    for (int r = 0; r < 8; ++r){ eA[r] = __expf(dA[r]-mn); eB[r] = __expf(dB[r]-mn); ps += eA[r]+eB[r]; }
    ps += __shfl_xor(ps, 16, 32);
    l = l*alpha + ps; m = mn;
    #pragma unroll
    for (int t = 0; t < 8; ++t)
      #pragma unroll
      for (int r = 0; r < 8; ++r) acc[t][r] *= alpha;
    unsigned pA[4], pB[4];
    #pragma unroll
    for (int j = 0; j < 4; ++j){ pA[j] = pack2(eA[2*j], eA[2*j+1]); pB[j] = pack2(eB[2*j], eB[2*j+1]); }
    v8u ub;
    #pragma unroll
    for (int j = 0; j < 4; ++j){
      unsigned pAx = __shfl_xor(pA[j], 16, 32);
      unsigned pBx = __shfl_xor(pB[j], 16, 32);
      ub[j]   = low ? pA[j] : pBx;             // keys 0..7  | 16..23
      ub[j+4] = low ? pAx   : pB[j];           // keys 8..15 | 24..31
    }
    return __builtin_bit_cast(v16h, ub);
  };

  v16h bq[8];                                 // two Q^T B-operand tiles, resident
  #pragma unroll
  for (int cc = 0; cc < 4; ++cc){
    bq[cc]   = load_b(Qb, q0,      CH, cc*32);
    bq[4+cc] = load_b(Qb, q0+16,   CH, cc*32);
  }

  v8f oacc[16];                               // O^T accumulators: 32q x 128ch
  #pragma unroll
  for (int t = 0; t < 16; ++t) oacc[t] = zero8();
  float m0 = -1e30f, l0 = 0.0f, m1 = -1e30f, l1 = 0.0f;

  issue(0, 0);
  for (int j0 = 0; j0 < HW; j0 += 32){
    int bi = (j0 >> 5) & 1;
    wait_async0();                 // own wave's DMA into buf bi landed
    __syncthreads();               // whole workgroup's DMA done; prior readers drained
    if (j0 + 32 < HW) issue(bi ^ 1, j0 + 32);   // prefetch next block during compute

    const _Float16* kS = &sK[bi][0];
    const _Float16* vS = &sV[bi][0];

    // S^T = K_tile x Q^T for both query tiles; every K A-tile feeds 2 WMMAs
    v8f d00 = zero8(), d01 = zero8(), d10 = zero8(), d11 = zero8();
    #pragma unroll
    for (int cc = 0; cc < 4; ++cc){
      v16h ka = load_a(kS, 0, SK_LD, cc*32);
      d00 = wmma_f16(ka, bq[cc],   d00);
      d10 = wmma_f16(ka, bq[4+cc], d10);
    }
    #pragma unroll
    for (int cc = 0; cc < 4; ++cc){
      v16h ka = load_a(kS, 16, SK_LD, cc*32);
      d01 = wmma_f16(ka, bq[cc],   d01);
      d11 = wmma_f16(ka, bq[4+cc], d11);
    }

    v16h bp0 = softmax_pack(d00, d01, m0, l0, &oacc[0]);
    v16h bp1 = softmax_pack(d10, d11, m1, l1, &oacc[8]);

    // O^T += V^T x P^T; every V A-tile feeds 2 WMMAs
    #pragma unroll
    for (int t = 0; t < 8; ++t){
      v16h va = load_a(vS, t*16, SV_LD, 0);
      oacc[t]   = wmma_f16(va, bp0, oacc[t]);
      oacc[8+t] = wmma_f16(va, bp1, oacc[8+t]);
    }
  }

  _Float16* Ob = O + (size_t)b*HW*CH;
  float inv0 = 1.0f / l0, inv1 = 1.0f / l1;
  int tok0 = q0 + r16, tok1 = q0 + 16 + r16;
  #pragma unroll
  for (int t = 0; t < 8; ++t)
    #pragma unroll
    for (int r = 0; r < 8; ++r){
      int c = t*16 + r + (low ? 0 : 8);
      Ob[(size_t)tok0*CH + c] = (_Float16)(oacc[t][r]*inv0);
      Ob[(size_t)tok1*CH + c] = (_Float16)(oacc[8+t][r]*inv1);
    }
}

// ---------------- proj GEMM + bias + residual -------------------------------
__global__ void proj_kernel(const _Float16* __restrict__ O, const _Float16* __restrict__ pw16,
                            const float* __restrict__ projb, const float* __restrict__ x,
                            float* __restrict__ out){
  int w = blockIdx.x*8 + (threadIdx.x >> 5);
  int b = w >> 8; int t0 = (w & 255) * 16;
  int lane = threadIdx.x & 31; bool low = lane < 16; int r16 = lane & 15;
  const _Float16* Bsrc = O + (size_t)b*HW*CH;
  v16h bo[4];
  #pragma unroll
  for (int cc = 0; cc < 4; ++cc) bo[cc] = load_b(Bsrc, t0, CH, cc*32);
  int token = t0 + r16;
  for (int ot = 0; ot < 8; ++ot){
    v16h a[4];                               // hoist: one clause of 8 loads, one wait
    #pragma unroll
    for (int cc = 0; cc < 4; ++cc) a[cc] = load_a(pw16, ot*16, CH, cc*32);
    v8f acc = zero8();
    #pragma unroll
    for (int cc = 0; cc < 4; ++cc) acc = wmma_f16(a[cc], bo[cc], acc);
    #pragma unroll
    for (int r = 0; r < 8; ++r){
      int o = ot*16 + r + (low ? 0 : 8);
      size_t idx = ((size_t)b*CH + o)*HW + token;
      out[idx] = x[idx] + acc[r] + projb[o];
    }
  }
}

extern "C" void kernel_launch(void* const* d_in, const int* in_sizes, int n_in,
                              void* d_out, int out_size, void* d_ws, size_t ws_size,
                              hipStream_t stream){
  const float* x      = (const float*)d_in[0];
  const float* norm_w = (const float*)d_in[1];
  const float* norm_b = (const float*)d_in[2];
  const float* qkv_w  = (const float*)d_in[3];
  const float* qkv_b  = (const float*)d_in[4];
  const float* proj_w = (const float*)d_in[5];
  const float* proj_b = (const float*)d_in[6];
  float* out = (float*)d_out;

  char* ws = (char*)d_ws;
  size_t off = 0;
  const size_t tensor_h = (size_t)4*HW*CH*sizeof(_Float16);   // 4 MB each
  float*    stats = (float*)(ws+off);    off += 256;
  _Float16* xnt   = (_Float16*)(ws+off); off += tensor_h;
  _Float16* qw16  = (_Float16*)(ws+off); off += (size_t)384*CH*2;
  _Float16* pw16  = (_Float16*)(ws+off); off += (size_t)CH*CH*2;
  _Float16* Qws   = (_Float16*)(ws+off); off += tensor_h;
  _Float16* Kws   = (_Float16*)(ws+off); off += tensor_h;
  _Float16* Vtws  = (_Float16*)(ws+off); off += tensor_h;
  _Float16* Ows   = (_Float16*)(ws+off); off += tensor_h;

  gn_stats_kernel<<<32, 256, 0, stream>>>(x, stats);
  gn_norm_kernel<<<512, 256, 0, stream>>>(x, norm_w, norm_b, stats, xnt);
  cvt_w_kernel<<<256, 256, 0, stream>>>(qkv_w, proj_w, qw16, pw16);
  qkv_gemm_kernel<<<128, 256, 0, stream>>>(xnt, qw16, qkv_b, Qws, Kws, Vtws);
  attn_kernel<<<128, 128, 0, stream>>>(Qws, Kws, Vtws, Ows);
  proj_kernel<<<128, 256, 0, stream>>>(Ows, pw16, proj_b, x, out);
}
